// KMEANS_67980742361657
// MI455X (gfx1250) — matrix operand order repaired and checked by reading, not compiled, not run on hardware
//
#include <hip/hip_runtime.h>
#include <hip/hip_bf16.h>

typedef __attribute__((ext_vector_type(16))) __bf16 v16bf;
typedef __attribute__((ext_vector_type(8)))  float  v8f;
typedef __attribute__((ext_vector_type(4)))  unsigned int v4u;

#define N_PTS 65536
#define DIM   512
#define KC    1024
#define NCLS  10
#define MB    64     // rows per workgroup
#define SP    520    // padded LDS row stride (bf16 elems): 260 dwords -> conflict-free b128

union BFrag { v16bf v; v4u u[2]; };

// ---------------------------------------------------------------- init counts
__global__ void zero_counts(int* __restrict__ counts) {
  int i = blockIdx.x * 256 + threadIdx.x;
  if (i < KC * NCLS) counts[i] = 0;
}

// ------------------------------------------------- centers fp32 -> bf16 hi/lo
__global__ __launch_bounds__(256) void prep_centers(
    const float* __restrict__ c, __bf16* __restrict__ chi,
    __bf16* __restrict__ clo, float* __restrict__ c2) {
  __shared__ float red[256];
  const int k = blockIdx.x, t = threadIdx.x;
  float ss = 0.f;
  for (int i = t; i < DIM; i += 256) {
    float v = c[(size_t)k * DIM + i];
    ss += v * v;
    __bf16 h = (__bf16)v;
    chi[(size_t)k * DIM + i] = h;
    clo[(size_t)k * DIM + i] = (__bf16)(v - (float)h);
  }
  red[t] = ss;
  __syncthreads();
  for (int o = 128; o > 0; o >>= 1) {
    if (t < o) red[t] += red[t + o];
    __syncthreads();
  }
  if (t == 0) c2[k] = red[0];
}

// ------------------------------------------------------- main GEMM + argmin
__global__ __launch_bounds__(256) void kmeans_main(
    const float* __restrict__ x, const int* __restrict__ y,
    const v4u* __restrict__ chi, const v4u* __restrict__ clo,
    const float* __restrict__ c2, int* __restrict__ counts,
    float* __restrict__ loss_part) {
  extern __shared__ char smem[];
  __bf16* sHi = (__bf16*)smem;                                   // MB*SP
  __bf16* sLo = sHi + MB * SP;                                   // MB*SP
  unsigned long long* rowMin = (unsigned long long*)(sLo + MB * SP); // MB
  float* red = (float*)(rowMin + MB);                            // 256

  const int tid = threadIdx.x;
  const int row0 = blockIdx.x * MB;

  // Stage x tile into LDS as bf16 hi/lo; accumulate sum of squares.
  float sumsq = 0.f;
#pragma unroll 4
  for (int i = 0; i < (MB * DIM) / 256; ++i) {
    int e = i * 256 + tid;
    int r = e >> 9, cc = e & (DIM - 1);
    float v = x[(size_t)(row0 + r) * DIM + cc];
    sumsq += v * v;
    __bf16 h = (__bf16)v;
    sHi[r * SP + cc] = h;
    sLo[r * SP + cc] = (__bf16)(v - (float)h);
  }
  if (tid < MB) rowMin[tid] = ~0ULL;
  __syncthreads();

  const int lane = tid & 31;
  const int wave = tid >> 5;
  const int rg = wave >> 2;   // row group: rows rg*32 .. rg*32+31
  const int cs = wave & 3;    // column strip
  const int l16 = lane >> 4;  // upper half-wave flag
  const int ln = lane & 15;

  unsigned long long rmin[16];
#pragma unroll
  for (int s = 0; s < 16; ++s) rmin[s] = ~0ULL;

#pragma unroll 1
  for (int it = 0; it < 16; ++it) {
    const int n0 = it * 64 + cs * 16;
    const int nb = n0 + ln;  // this lane's center (column) index
    const float c2n = c2[nb];
    v8f acc0 = {}; v8f acc1 = {};
    const int rowA0 = rg * 32 + ln;

    // Bounded unroll: keep A fragments as short-lived ds_load_b128 inside the
    // loop (a full unroll makes the compiler hoist ~512 VGPRs of A fragments
    // out of the `it` loop and spill to scratch).
#pragma unroll 2
    for (int kb = 0; kb < DIM; kb += 32) {
      // B fragments: lane ln holds column nb; lanes 0-15 -> K[kb..kb+15],
      // lanes 16-31 -> K[kb+16..kb+31]; two b128 loads each.
      const int kB = kb + (l16 ? 16 : 0);
      const size_t bidx = ((size_t)nb * DIM + kB) >> 3;
      BFrag bh, bl;
      bh.u[0] = chi[bidx]; bh.u[1] = chi[bidx + 1];
      bl.u[0] = clo[bidx]; bl.u[1] = clo[bidx + 1];

      // A fragments: lanes 0-15 -> K[kb..kb+15],
      // lanes 16-31 -> K[kb+16..kb+31] (same k order as B).
      const int kA = kb + (l16 ? 16 : 0);
      const int a0 = rowA0 * SP + kA;
      const int a1 = (rowA0 + 16) * SP + kA;
      BFrag ah0, al0, ah1, al1;
      ah0.u[0] = *(const v4u*)(sHi + a0); ah0.u[1] = *(const v4u*)(sHi + a0 + 8);
      al0.u[0] = *(const v4u*)(sLo + a0); al0.u[1] = *(const v4u*)(sLo + a0 + 8);
      ah1.u[0] = *(const v4u*)(sHi + a1); ah1.u[1] = *(const v4u*)(sHi + a1 + 8);
      al1.u[0] = *(const v4u*)(sLo + a1); al1.u[1] = *(const v4u*)(sLo + a1 + 8);

      // hi*hi + hi*lo + lo*hi split for ~fp32 accuracy, f32 accumulate.
      acc0 = __builtin_amdgcn_wmma_f32_16x16x32_bf16(false, ah0.v, false, bh.v, (short)0, acc0, false, false);
      acc0 = __builtin_amdgcn_wmma_f32_16x16x32_bf16(false, ah0.v, false, bl.v, (short)0, acc0, false, false);
      acc0 = __builtin_amdgcn_wmma_f32_16x16x32_bf16(false, al0.v, false, bh.v, (short)0, acc0, false, false);
      acc1 = __builtin_amdgcn_wmma_f32_16x16x32_bf16(false, ah1.v, false, bh.v, (short)0, acc1, false, false);
      acc1 = __builtin_amdgcn_wmma_f32_16x16x32_bf16(false, ah1.v, false, bl.v, (short)0, acc1, false, false);
      acc1 = __builtin_amdgcn_wmma_f32_16x16x32_bf16(false, al1.v, false, bh.v, (short)0, acc1, false, false);
    }

    // Fold into running per-row min with monotonic float key | center idx.
#pragma unroll
    for (int r = 0; r < 8; ++r) {
      float d0 = c2n - 2.0f * acc0[r];
      unsigned int fb0 = __float_as_uint(d0);
      unsigned int m0 = (fb0 & 0x80000000u) ? ~fb0 : (fb0 | 0x80000000u);
      unsigned long long k0 = ((unsigned long long)m0 << 32) | (unsigned int)nb;
      if (k0 < rmin[r]) rmin[r] = k0;
      float d1 = c2n - 2.0f * acc1[r];
      unsigned int fb1 = __float_as_uint(d1);
      unsigned int m1 = (fb1 & 0x80000000u) ? ~fb1 : (fb1 | 0x80000000u);
      unsigned long long k1 = ((unsigned long long)m1 << 32) | (unsigned int)nb;
      if (k1 < rmin[8 + r]) rmin[8 + r] = k1;
    }
  }

  // Merge across lanes/waves via LDS 64-bit min atomics.
#pragma unroll
  for (int r = 0; r < 8; ++r) {
    int rl0 = rg * 32 + r + l16 * 8;  // C layout: VGPR r, lanes>=16 -> row+8
    atomicMin(&rowMin[rl0], rmin[r]);
    atomicMin(&rowMin[rl0 + 16], rmin[8 + r]);
  }
  __syncthreads();

  // Reduce sum(x^2) over the workgroup (fixed order => deterministic).
  red[tid] = sumsq;
  __syncthreads();
  for (int o = 128; o > 0; o >>= 1) {
    if (tid < o) red[tid] += red[tid + o];
    __syncthreads();
  }
  const float total_x2 = red[0];
  __syncthreads();

  // Decode per-row winners: loss contribution + cluster/label counts.
  float dmin = 0.f;
  if (tid < MB) {
    unsigned long long key = rowMin[tid];
    unsigned int m = (unsigned int)(key >> 32);
    unsigned int fb = (m & 0x80000000u) ? (m & 0x7FFFFFFFu) : ~m;
    dmin = __uint_as_float(fb);
    unsigned int idx = (unsigned int)(key & 0xFFFFFFFFu);
    atomicAdd(&counts[idx * NCLS + y[row0 + tid]], 1);
  }
  red[tid] = dmin;
  __syncthreads();
  for (int o = 128; o > 0; o >>= 1) {
    if (tid < o) red[tid] += red[tid + o];
    __syncthreads();
  }
  if (tid == 0) loss_part[blockIdx.x] = total_x2 + red[0];
}

// ---------------------------------------------------------------- finalize
__global__ __launch_bounds__(1024) void finalize(
    const int* __restrict__ counts, const float* __restrict__ loss_part,
    float* __restrict__ out) {
  __shared__ int ri[1024];
  __shared__ float rf[1024];
  const int t = threadIdx.x;
  int mx = 0;
#pragma unroll
  for (int c = 0; c < NCLS; ++c) mx = max(mx, counts[t * NCLS + c]);
  ri[t] = mx;
  rf[t] = loss_part[t];
  __syncthreads();
  for (int o = 512; o > 0; o >>= 1) {
    if (t < o) { ri[t] += ri[t + o]; rf[t] += rf[t + o]; }
    __syncthreads();
  }
  if (t == 0) {
    out[0] = rf[0];
    out[1] = (float)ri[0] / (float)N_PTS;
  }
}

// ---------------------------------------------------------------- launch
extern "C" void kernel_launch(void* const* d_in, const int* in_sizes, int n_in,
                              void* d_out, int out_size, void* d_ws, size_t ws_size,
                              hipStream_t stream) {
  const float* x = (const float*)d_in[0];
  const float* centers = (const float*)d_in[1];
  const int* y = (const int*)d_in[2];
  float* out = (float*)d_out;

  char* ws = (char*)d_ws;
  __bf16* chi = (__bf16*)ws;                         // 1 MB
  __bf16* clo = (__bf16*)(ws + 1048576);             // 1 MB
  float* c2 = (float*)(ws + 2097152);                // 4 KB
  int* counts = (int*)(ws + 2101248);                // 40 KB
  float* loss_part = (float*)(ws + 2142208);         // 4 KB

  zero_counts<<<(KC * NCLS + 255) / 256, 256, 0, stream>>>(counts);
  prep_centers<<<KC, 256, 0, stream>>>(centers, chi, clo, c2);
  const size_t smem = (size_t)MB * SP * 2 * sizeof(__bf16)   // x hi/lo tiles
                    + (size_t)MB * 8                         // rowMin
                    + 256 * 4;                               // red
  kmeans_main<<<N_PTS / MB, 256, smem, stream>>>(
      x, y, (const v4u*)chi, (const v4u*)clo, c2, counts, loss_part);
  finalize<<<1, 1024, 0, stream>>>(counts, loss_part, out);
}
